// PortableMambaMixer_9835475108384
// MI455X (gfx1250) — matrix operand order repaired and checked by reading, not compiled, not run on hardware
//
#include <hip/hip_runtime.h>

// ---------------- problem constants (match reference) ----------------
#define D_MODEL 1024
#define D_INNER 2048
#define D_STATE 16
#define DT_RANK 64
#define BB      2
#define SS      2048
#define ROWS    (BB * SS)          // 4096 token rows
#define NCH     16                 // scan chunks per sequence
#define CHL     (SS / NCH)         // 128 steps per chunk

typedef __attribute__((ext_vector_type(16))) __bf16       v16bf;
typedef __attribute__((ext_vector_type(8)))  __bf16       v8bf;
typedef __attribute__((ext_vector_type(8)))  float        v8f;
typedef __attribute__((ext_vector_type(4)))  float        v4f;
typedef __attribute__((ext_vector_type(4)))  unsigned int v4u;

__device__ __forceinline__ float silu_f(float v) {
    return v * (1.0f / (1.0f + __expf(-v)));
}
__device__ __forceinline__ float softplus_f(float v) {
    return (v > 20.0f) ? v : log1pf(__expf(v));
}
__device__ __forceinline__ unsigned int pack_bf16(float lo, float hi) {
    unsigned short l = __builtin_bit_cast(unsigned short, (__bf16)lo);
    unsigned short h = __builtin_bit_cast(unsigned short, (__bf16)hi);
    return (unsigned int)l | ((unsigned int)h << 16);
}

// =====================================================================
// LDS-tiled GEMM with double-buffered (ping-pong) LDS:
//   C[M,N] = epilogue( A[M,K] @ B[K,N] ), fp32 memory, bf16 WMMA
//   fragments, fp32 accumulation.
// Block tile 128x128x32, 256 threads = 8 waves (2x4), wave tile 64x32
// = 4x2 WMMA tiles -> 8 v_wmma per K-step per wave.
// Pipeline per K-step: issue global loads for tile i+1, compute tile i
// from LDS buffer (i&1), convert+store tile i+1 into buffer (i+1)&1,
// ONE barrier. Global-load latency hides under 8 WMMAs + ds loads.
// LDS rows padded to 20 dwords (16 data) so 16-lane b128 fragment reads
// hit 16 distinct banks (gcd(20,64)=4, period 16). 2x20KB = 40KB LDS.
// Requires: M%128==0, N%128==0, K%32==0 (true for all 3 big GEMMs).
// EPI: 0 = none, 1 = +bias[n] then softplus.
// =====================================================================
#define BM 128
#define BN 128
#define BK 32
#define LDSA_STRIDE 20   // dwords per A row (32 bf16 = 16 dwords + pad)
#define LDSB_STRIDE 20   // dwords per B^T row

template <int EPI>
__launch_bounds__(256)
__global__ void gemm_tiled_bf16(const float* __restrict__ A, int lda,
                                const float* __restrict__ B, int ldb,
                                float*       __restrict__ C, int ldc,
                                int K, int tilesNb,
                                const float* __restrict__ bias)
{
    __shared__ unsigned int ldsA[2][BM * LDSA_STRIDE];  // 2 x 10 KB
    __shared__ unsigned int ldsB[2][BN * LDSB_STRIDE];  // 2 x 10 KB

    const int tid   = threadIdx.x;
    const int lane  = tid & 31;
    const int wave  = tid >> 5;
    const int waveM = wave >> 2;       // 0..1  -> row offset waveM*64
    const int waveN = wave & 3;        // 0..3  -> col offset waveN*32
    const int half  = lane >> 4;       // 0: lanes 0-15, 1: lanes 16-31
    const int l16   = lane & 15;

    const int bm = blockIdx.x / tilesNb;
    const int bn = blockIdx.x - bm * tilesNb;
    const int m0 = bm * BM;
    const int n0 = bn * BN;

    // staging coordinates
    const int sm  = tid >> 1;          // A row 0..127
    const int sko = (tid & 1) * 16;    // A k offset {0,16}
    const int snb = tid & 127;         // B col 0..127
    const int skb = (tid >> 7) * 16;   // B k base {0,16}

    const float* Abase = A + (size_t)(m0 + sm) * lda + sko;
    const float* Bbase = B + (n0 + snb);

    v8f acc[4][2];
#pragma unroll
    for (int mt = 0; mt < 4; ++mt)
#pragma unroll
        for (int nt = 0; nt < 2; ++nt) acc[mt][nt] = (v8f){};

    v4f   areg[4];
    float breg[16];

    auto load_global = [&](int kk) {
        const v4f* ap = (const v4f*)(Abase + kk);
#pragma unroll
        for (int j = 0; j < 4; ++j) areg[j] = ap[j];
#pragma unroll
        for (int j = 0; j < 16; ++j)
            breg[j] = Bbase[(size_t)(kk + skb + j) * ldb];
    };

    auto store_lds = [&](int buf) {
        unsigned int* pa = &ldsA[buf][0] + sm * LDSA_STRIDE + (sko >> 1);
#pragma unroll
        for (int j = 0; j < 4; ++j) {
            pa[2 * j]     = pack_bf16(areg[j][0], areg[j][1]);
            pa[2 * j + 1] = pack_bf16(areg[j][2], areg[j][3]);
        }
        unsigned int* pb = &ldsB[buf][0] + snb * LDSB_STRIDE + (skb >> 1);
#pragma unroll
        for (int j = 0; j < 8; ++j)
            pb[j] = pack_bf16(breg[2 * j], breg[2 * j + 1]);
    };

    auto compute = [&](int buf) {
        v16bf afr[4];
#pragma unroll
        for (int mt = 0; mt < 4; ++mt) {
            const int m = waveM * 64 + mt * 16 + l16;
            const v4u* p = (const v4u*)(&ldsA[buf][0] + m * LDSA_STRIDE);
            v8bf lo = __builtin_bit_cast(v8bf, p[half * 2]);     // k = half*16..+7
            v8bf hi = __builtin_bit_cast(v8bf, p[half * 2 + 1]); // k = half*16+8..
            afr[mt] = __builtin_shufflevector(lo, hi, 0, 1, 2, 3, 4, 5, 6, 7,
                                              8, 9, 10, 11, 12, 13, 14, 15);
        }
        v16bf bfr[2];
#pragma unroll
        for (int nt = 0; nt < 2; ++nt) {
            const int n = waveN * 32 + nt * 16 + l16;
            const v4u* p = (const v4u*)(&ldsB[buf][0] + n * LDSB_STRIDE);
            v8bf lo = __builtin_bit_cast(v8bf, p[half * 2]);     // k=half*16..
            v8bf hi = __builtin_bit_cast(v8bf, p[half * 2 + 1]); // ..+15
            bfr[nt] = __builtin_shufflevector(lo, hi, 0, 1, 2, 3, 4, 5, 6, 7,
                                              8, 9, 10, 11, 12, 13, 14, 15);
        }
#pragma unroll
        for (int mt = 0; mt < 4; ++mt)
#pragma unroll
            for (int nt = 0; nt < 2; ++nt)
                acc[mt][nt] = __builtin_amdgcn_wmma_f32_16x16x32_bf16(
                    false, afr[mt], false, bfr[nt], (short)0, acc[mt][nt],
                    false, false);
    };

    // ---- pipeline prologue: stage tile 0 into buffer 0 ----
    load_global(0);
    store_lds(0);
    __syncthreads();

    const int nk = K / BK;
    for (int i = 0; i < nk; ++i) {
        const int cur = i & 1;
        const bool more = (i + 1 < nk);
        if (more) {
            load_global((i + 1) * BK);          // overlaps with compute below
            if (i + 2 < nk) {
                __builtin_prefetch(Abase + (i + 2) * BK, 0, 1);
                __builtin_prefetch(Bbase + (size_t)((i + 2) * BK + skb) * ldb, 0, 1);
            }
        }
        compute(cur);                           // 12 ds_load_b128 + 8 v_wmma
        if (more) {
            store_lds(cur ^ 1);                 // other buffer: no wait needed
            __syncthreads();                    // one barrier per K-step
        }
    }

    // ---- epilogue ----
#pragma unroll
    for (int mt = 0; mt < 4; ++mt) {
#pragma unroll
        for (int nt = 0; nt < 2; ++nt) {
            const int row = m0 + waveM * 64 + mt * 16 + half * 8;
            const int col = n0 + waveN * 32 + nt * 16 + l16;
            const float bval = (EPI == 1) ? bias[col] : 0.0f;
            float* Cp = C + (size_t)row * ldc + col;
#pragma unroll
            for (int v = 0; v < 8; ++v) {
                float val = acc[mt][nt][v];
                if (EPI == 1) val = softplus_f(val + bval);
                Cp[(size_t)v * ldc] = val;
            }
        }
    }
}

// =====================================================================
// Simple wave-per-16x16-tile GEMM (for the N=96 Wx projection only).
// =====================================================================
__launch_bounds__(128)
__global__ void gemm_simple_bf16(const float* __restrict__ A, int lda,
                                 const float* __restrict__ B, int ldb,
                                 float*       __restrict__ C, int ldc,
                                 int K, int tilesN)
{
    const int lane = threadIdx.x & 31;
    const int wave = threadIdx.x >> 5;
    const int tile = blockIdx.x * 4 + wave;
    const int tm   = tile / tilesN;
    const int tn   = tile - tm * tilesN;
    const int half = lane >> 4;
    const int l16  = lane & 15;

    const float* Arow = A + (size_t)(tm * 16 + l16) * lda;
    const float* Bcol = B + (tn * 16 + l16);

    v8f acc = {};
    for (int k0 = 0; k0 < K; k0 += 32) {
        v16bf av, bv;
#pragma unroll
        for (int j = 0; j < 8; ++j) {
            av[j]     = (__bf16)Arow[k0 + half * 16 + j];
            av[j + 8] = (__bf16)Arow[k0 + half * 16 + 8 + j];
        }
#pragma unroll
        for (int e = 0; e < 16; ++e)
            bv[e] = (__bf16)Bcol[(size_t)(k0 + half * 16 + e) * ldb];
        acc = __builtin_amdgcn_wmma_f32_16x16x32_bf16(
                  false, av, false, bv, (short)0, acc, false, false);
    }
    float* Cout = C + (size_t)(tm * 16 + half * 8) * ldc + (tn * 16 + l16);
#pragma unroll
    for (int v = 0; v < 8; ++v) Cout[(size_t)v * ldc] = acc[v];
}

// =====================================================================
// Causal depthwise conv (K=4) + bias + SiLU on the x-branch of xz.
// =====================================================================
__launch_bounds__(256)
__global__ void conv_silu_kernel(const float* __restrict__ xz,
                                 const float* __restrict__ Wconv,
                                 const float* __restrict__ bconv,
                                 float*       __restrict__ xc)
{
    const int idx = blockIdx.x * 256 + threadIdx.x;   // ROWS*D_INNER
    const int d   = idx & (D_INNER - 1);
    const int row = idx >> 11;
    const int s   = row & (SS - 1);

    const float w0 = Wconv[d * 4 + 0], w1 = Wconv[d * 4 + 1];
    const float w2 = Wconv[d * 4 + 2], w3 = Wconv[d * 4 + 3];

    const size_t stride = 2 * D_INNER;
    const size_t base   = (size_t)row * stride + d;

    float acc = bconv[d];
    acc += (s >= 3 ? xz[base - 3 * stride] : 0.0f) * w0;
    acc += (s >= 2 ? xz[base - 2 * stride] : 0.0f) * w1;
    acc += (s >= 1 ? xz[base - 1 * stride] : 0.0f) * w2;
    acc += xz[base] * w3;

    xc[(size_t)row * D_INNER + d] = silu_f(acc);
}

// =====================================================================
// Chunked selective scan (linear recurrence split into 16 chunks):
// pass1 scans chunks with zero init (records P = prod dA, S = partial
// final state), pass2 does the 16-step cross-chunk recurrence, pass3
// adds the correction C_t.(cum_t*H_c) and fuses the SiLU gate.
// =====================================================================
__launch_bounds__(256)
__global__ void scan_pass1(const float* __restrict__ xc,
                           const float* __restrict__ delta,
                           const float* __restrict__ params,
                           const float* __restrict__ A_log,
                           const float* __restrict__ Dp,
                           float* __restrict__ y_part,
                           float* __restrict__ Pc,
                           float* __restrict__ Sc)
{
    const int gid = blockIdx.x * 256 + threadIdx.x;   // BB*NCH*D_INNER
    const int d   = gid & (D_INNER - 1);
    const int c   = (gid >> 11) & (NCH - 1);
    const int b   = gid >> 15;

    float a[D_STATE];
#pragma unroll
    for (int n = 0; n < D_STATE; ++n) a[n] = -__expf(A_log[d * D_STATE + n]);

    float st[D_STATE], cum[D_STATE];
#pragma unroll
    for (int n = 0; n < D_STATE; ++n) { st[n] = 0.0f; cum[n] = 1.0f; }

    const float dpv = Dp[d];

    for (int tt = 0; tt < CHL; ++tt) {
        const size_t row = (size_t)b * SS + c * CHL + tt;
        const float  dt  = delta[row * D_INNER + d];
        const float  u   = xc[row * D_INNER + d];
        const float  dtu = dt * u;
        const float* p   = params + row * 96;

        float y = u * dpv;
#pragma unroll
        for (int n = 0; n < D_STATE; ++n) {
            const float dA = __expf(dt * a[n]);
            st[n]  = dA * st[n] + dtu * p[DT_RANK + n];
            cum[n] *= dA;
            y += st[n] * p[DT_RANK + D_STATE + n];
        }
        y_part[row * D_INNER + d] = y;
    }
#pragma unroll
    for (int n = 0; n < D_STATE; ++n) {
        const size_t idx = ((size_t)(b * NCH + c) * D_STATE + n) * D_INNER + d;
        Sc[idx] = st[n];
        Pc[idx] = cum[n];
    }
}

__launch_bounds__(256)
__global__ void scan_pass2(const float* __restrict__ Pc,
                           const float* __restrict__ Sc,
                           float* __restrict__ H)
{
    const int gid = blockIdx.x * 256 + threadIdx.x;   // BB*D_INNER
    const int d   = gid & (D_INNER - 1);
    const int b   = gid >> 11;

    float h[D_STATE];
#pragma unroll
    for (int n = 0; n < D_STATE; ++n) h[n] = 0.0f;

    for (int c = 0; c < NCH; ++c) {
#pragma unroll
        for (int n = 0; n < D_STATE; ++n) {
            const size_t idx = ((size_t)(b * NCH + c) * D_STATE + n) * D_INNER + d;
            H[idx] = h[n];                       // true init state of chunk c
            h[n] = Sc[idx] + Pc[idx] * h[n];     // true final state of chunk c
        }
    }
}

__launch_bounds__(256)
__global__ void scan_pass3_gate(const float* __restrict__ delta,
                                const float* __restrict__ params,
                                const float* __restrict__ A_log,
                                const float* __restrict__ y_part,
                                const float* __restrict__ H,
                                const float* __restrict__ xz,
                                float* __restrict__ yg)
{
    const int gid = blockIdx.x * 256 + threadIdx.x;   // BB*NCH*D_INNER
    const int d   = gid & (D_INNER - 1);
    const int c   = (gid >> 11) & (NCH - 1);
    const int b   = gid >> 15;

    float a[D_STATE], h[D_STATE];
#pragma unroll
    for (int n = 0; n < D_STATE; ++n) {
        a[n] = -__expf(A_log[d * D_STATE + n]);
        h[n] = H[((size_t)(b * NCH + c) * D_STATE + n) * D_INNER + d];
    }
    const bool has_corr = (c != 0);   // chunk 0 has zero incoming state

    for (int tt = 0; tt < CHL; ++tt) {
        const size_t row = (size_t)b * SS + c * CHL + tt;
        float y = y_part[row * D_INNER + d];
        if (has_corr) {
            const float  dt = delta[row * D_INNER + d];
            const float* p  = params + row * 96;
            float ycorr = 0.0f;
#pragma unroll
            for (int n = 0; n < D_STATE; ++n) {
                h[n] *= __expf(dt * a[n]);            // cum_t * H_c
                ycorr += h[n] * p[DT_RANK + D_STATE + n];
            }
            y += ycorr;
        }
        const float z = xz[row * (2 * D_INNER) + D_INNER + d];
        yg[row * D_INNER + d] = y * silu_f(z);
    }
}

// =====================================================================
extern "C" void kernel_launch(void* const* d_in, const int* in_sizes, int n_in,
                              void* d_out, int out_size, void* d_ws, size_t ws_size,
                              hipStream_t stream)
{
    const float* x     = (const float*)d_in[0];  // [4096,1024]
    const float* Win   = (const float*)d_in[1];  // [1024,4096]
    const float* Wconv = (const float*)d_in[2];  // [2048,1,4]
    const float* bconv = (const float*)d_in[3];  // [2048]
    const float* Wx    = (const float*)d_in[4];  // [2048,96]
    const float* Wdt   = (const float*)d_in[5];  // [64,2048]
    const float* bdt   = (const float*)d_in[6];  // [2048]
    const float* A_log = (const float*)d_in[7];  // [2048,16]
    const float* Dp    = (const float*)d_in[8];  // [2048]
    const float* Wout  = (const float*)d_in[9];  // [2048,1024]
    float* out = (float*)d_out;                  // [4096,1024]

    char* ws = (char*)d_ws;
    float* xz     = (float*)(ws);                          //  64 MB [4096,4096]
    float* xc     = (float*)(ws + ((size_t)64  << 20));    //  32 MB [4096,2048]
    float* params = (float*)(ws + ((size_t)96  << 20));    // 1.5 MB [4096,96]
    float* delta  = (float*)(ws + ((size_t)98  << 20));    //  32 MB [4096,2048]
    float* y_part = (float*)(ws + ((size_t)130 << 20));    //  32 MB [4096,2048]
    float* Pc     = (float*)(ws + ((size_t)162 << 20));    //   8 MB
    float* Sc     = (float*)(ws + ((size_t)170 << 20));    //   8 MB
    float* H      = (float*)(ws + ((size_t)178 << 20));    //   8 MB
    float* yg     = delta;   // delta read+overwritten in-place by pass3 (safe)

    // 1) xz = x @ Win                       (4096x4096x1024)
    gemm_tiled_bf16<0><<<(ROWS / BM) * ((2 * D_INNER) / BN), 256, 0, stream>>>(
        x, D_MODEL, Win, 2 * D_INNER, xz, 2 * D_INNER, D_MODEL,
        (2 * D_INNER) / BN, nullptr);

    // 2) xc = silu(causal_dwconv(x_branch) + bconv)
    conv_silu_kernel<<<(ROWS * D_INNER) / 256, 256, 0, stream>>>(
        xz, Wconv, bconv, xc);

    // 3) params = xc @ Wx                   (4096x96x2048, N=96 -> simple)
    gemm_simple_bf16<<<(256 * 6) / 4, 128, 0, stream>>>(
        xc, D_INNER, Wx, 96, params, 96, D_INNER, 6);

    // 4) delta = softplus(params[:, :64] @ Wdt + bdt)   (4096x2048x64)
    gemm_tiled_bf16<1><<<(ROWS / BM) * (D_INNER / BN), 256, 0, stream>>>(
        params, 96, Wdt, D_INNER, delta, D_INNER, DT_RANK, D_INNER / BN, bdt);

    // 5) chunked selective scan
    scan_pass1<<<(BB * NCH * D_INNER) / 256, 256, 0, stream>>>(
        xc, delta, params, A_log, Dp, y_part, Pc, Sc);
    scan_pass2<<<(BB * D_INNER) / 256, 256, 0, stream>>>(Pc, Sc, H);
    scan_pass3_gate<<<(BB * NCH * D_INNER) / 256, 256, 0, stream>>>(
        delta, params, A_log, y_part, H, xz, yg);

    // 6) out = yg @ Wout                    (4096x1024x2048)
    gemm_tiled_bf16<0><<<(ROWS / BM) * (D_MODEL / BN), 256, 0, stream>>>(
        yg, D_INNER, Wout, D_MODEL, out, D_MODEL, D_INNER, D_MODEL / BN, nullptr);
}
